// Attention_8632884265040
// MI455X (gfx1250) — matrix operand order repaired and checked by reading, not compiled, not run on hardware
//
#include <hip/hip_runtime.h>
#include <hip/hip_bf16.h>
#include <stdint.h>

// ---- problem constants --------------------------------------------------
#define S_LEN   2048
#define DMODEL  1280
#define NHEADS  10
#define HDIM    128
#define BATCH   4
#define MROWS   (BATCH * S_LEN)      // 8192
#define N_QKV   (3 * DMODEL)         // 3840
#define BH      (BATCH * NHEADS)     // 40
#define ATT_SCALE 0.02795084971874737f   // 1280^-0.5 (reference uses D, not head_dim)

// ---- types --------------------------------------------------------------
typedef __bf16 bf16;
typedef bf16         v16bf __attribute__((ext_vector_type(16)));
typedef float        v8f   __attribute__((ext_vector_type(8)));
typedef unsigned int u32x4 __attribute__((ext_vector_type(4)));
typedef int          v4i_vs __attribute__((vector_size(16)));   // matches builtin proto

union FragAB { v16bf v; u32x4 q[2]; };
union FragC  { v8f   v; float f[8]; };

#if __has_builtin(__builtin_amdgcn_global_load_async_to_lds_b128)
#define HAVE_ASYNC_LDS 1
#endif

// 16-byte global -> LDS copy: async (ASYNCcnt) on CDNA5, sync fallback else.
__device__ __forceinline__ void async_cp16(unsigned short* lds, const unsigned short* g) {
#ifdef HAVE_ASYNC_LDS
  // prototype (from hipcc diagnostic): (int4 AS1*, int4 AS3*, imm offset, imm cpol)
  __builtin_amdgcn_global_load_async_to_lds_b128(
      (__attribute__((address_space(1))) v4i_vs*)const_cast<unsigned short*>(g),
      (__attribute__((address_space(3))) v4i_vs*)lds,
      /*offset=*/0, /*cpol=*/0);
#else
  *(u32x4*)lds = *(const u32x4*)g;
#endif
}

__device__ __forceinline__ void wait_async_all() {
#ifdef HAVE_ASYNC_LDS
  asm volatile("s_wait_asynccnt 0x0" ::: "memory");
#endif
}

__device__ __forceinline__ unsigned short f32_to_bf16(float f) {
  unsigned int u = __builtin_bit_cast(unsigned int, f);
  u = (u + 0x7FFFu + ((u >> 16) & 1u)) >> 16;   // round-to-nearest-even
  return (unsigned short)u;
}

// A-matrix 16x32 bf16 fragment (ISA 7.12.2): lanes 0-15 -> M=lane, K={0..7,16..23};
// lanes 16-31 -> M=lane-16, K={8..15,24..31}. Two 128-bit loads per lane.
__device__ __forceinline__ void load_afrag(FragAB& f, const unsigned short* a,
                                           int ld, int row0, int k0, int lane) {
  const int r    = lane & 15;
  const int koff = (lane & 16) ? 8 : 0;
  const unsigned short* p = a + (size_t)(row0 + r) * ld + k0 + koff;
  f.q[0] = *(const u32x4*)(p);        // K = koff .. koff+7
  f.q[1] = *(const u32x4*)(p + 16);   // K = koff+16 .. koff+23
}

// B-matrix 32x16 bf16 fragment from a row-major B^T source ([N][K]):
// lane n<16 holds column N=n, K={0..7,16..23}; lane n+16 holds K={8..15,24..31}.
__device__ __forceinline__ void load_bfrag_lds(FragAB& f, const unsigned short* tile,
                                               int ld, int n0, int k0, int lane) {
  const int n    = lane & 15;
  const int koff = (lane & 16) ? 8 : 0;
  const unsigned short* p = tile + (n0 + n) * ld + k0 + koff;
  f.q[0] = *(const u32x4*)(p);
  f.q[1] = *(const u32x4*)(p + 16);
}

__device__ __forceinline__ v8f wmma_bf16(v16bf a, v16bf b, v8f c) {
  // 8 args: (neg_a, A, neg_b, B, c_mod, C, reuse_a, reuse_b)
  return __builtin_amdgcn_wmma_f32_16x16x32_bf16(false, a, false, b, (short)0, c,
                                                 false, false);
}

// Stage a 64(N) x 32(K) bf16 B tile into LDS; 256 threads, one 16B chunk each.
__device__ __forceinline__ void stage_btile(unsigned short* dst /*[64][32]*/,
                                            const unsigned short* bt, int ld,
                                            int n0, int k0, int t) {
  const int row = t >> 2;          // 0..63
  const int off = (t & 3) * 8;     // 0,8,16,24
  async_cp16(dst + row * 32 + off, bt + (size_t)(n0 + row) * ld + k0 + off);
}

// ---- kernel 1: casts ----------------------------------------------------
__global__ void cast_bf16_kernel(const float* __restrict__ src,
                                 unsigned short* __restrict__ dst, int n) {
  int i = blockIdx.x * blockDim.x + threadIdx.x;
  if (i < n) dst[i] = f32_to_bf16(src[i]);
}

// src [R][C] fp32 -> dst [C][R] bf16
__global__ void transpose_cast_kernel(const float* __restrict__ src,
                                      unsigned short* __restrict__ dst,
                                      int R, int C) {
  long i = (long)blockIdx.x * blockDim.x + threadIdx.x;
  long total = (long)R * C;
  if (i < total) {
    int c = (int)(i / R);
    int r = (int)(i - (long)c * R);
    dst[i] = f32_to_bf16(src[(size_t)r * C + c]);
  }
}

// ---- kernel 2: QKV projection GEMM, de-interleaving epilogue ------------
// C[8192 x 3840] = xb[8192 x 1280] @ wqkvT^T ; wave computes 16x64; the
// shared 64x32 B tile is async-staged into LDS (double-buffered).
__global__ __launch_bounds__(256) void gemm_qkv_kernel(
    const unsigned short* __restrict__ xb, const unsigned short* __restrict__ wqkvT,
    unsigned short* __restrict__ qb, unsigned short* __restrict__ kb,
    unsigned short* __restrict__ vtb) {
  const int t    = threadIdx.x;
  const int lane = t & 31;
  const int wave = t >> 5;
  const int m0 = blockIdx.x * 128 + wave * 16;
  const int n0 = blockIdx.y * 64;

  __shared__ __align__(16) unsigned short btile[2][64 * 32];

  stage_btile(btile[0], wqkvT, DMODEL, n0, 0, t);

  FragC c[4] = {};
  for (int k0 = 0; k0 < DMODEL; k0 += 32) {
    const int buf = (k0 >> 5) & 1;
    wait_async_all();
    __syncthreads();                                    // tile[buf] complete
    if (k0 + 32 < DMODEL)
      stage_btile(btile[buf ^ 1], wqkvT, DMODEL, n0, k0 + 32, t);

    FragAB af; load_afrag(af, xb, DMODEL, m0, k0, lane);
#pragma unroll
    for (int nt = 0; nt < 4; ++nt) {
      FragAB bf; load_bfrag_lds(bf, btile[buf], 32, nt * 16, 0, lane);
      c[nt].v = wmma_bf16(af.v, bf.v, c[nt].v);
    }
    __syncthreads();                                    // done reading tile[buf]
  }

  // C layout: VGPR r holds M=r (lanes 0-15) / M=r+8 (lanes 16-31), N=lane&15.
  const int col   = lane & 15;
  const int rbase = (lane & 16) ? 8 : 0;
#pragma unroll
  for (int nt = 0; nt < 4; ++nt) {
#pragma unroll
    for (int r = 0; r < 8; ++r) {
      int m = m0 + rbase + r;
      int n = n0 + nt * 16 + col;
      int chunk = n / DMODEL;                  // 0:q 1:k 2:v
      int rem   = n - chunk * DMODEL;
      int head  = rem >> 7;
      int d     = rem & (HDIM - 1);
      int b     = m >> 11;                     // /2048
      int s     = m & (S_LEN - 1);
      size_t bh = (size_t)(b * NHEADS + head);
      unsigned short val = f32_to_bf16(c[nt].f[r]);
      if (chunk == 0)      qb[(bh * S_LEN + s) * HDIM + d] = val;
      else if (chunk == 1) kb[(bh * S_LEN + s) * HDIM + d] = val;
      else                 vtb[(bh * HDIM + d) * S_LEN + s] = val;   // V transposed
    }
  }
}

// ---- kernel 3: flash attention -----------------------------------------
// Block: 128 threads = 4 waves; each wave owns 16 query rows of one (b,h).
// The K (32x128) and V^T (128x32) tiles for each kv step are async-staged
// into LDS and shared by all 4 waves (double-buffered).
__global__ __launch_bounds__(128) void flash_attn_kernel(
    const unsigned short* __restrict__ qb, const unsigned short* __restrict__ kb,
    const unsigned short* __restrict__ vtb, unsigned short* __restrict__ ao) {
  const int t    = threadIdx.x;
  const int lane = t & 31;
  const int wave = t >> 5;
  const int bh   = blockIdx.y;
  const int b    = bh / NHEADS;
  const int h    = bh - b * NHEADS;
  const int q0   = blockIdx.x * 64 + wave * 16;

  const unsigned short* qp = qb  + (size_t)bh * S_LEN * HDIM;
  const unsigned short* kp = kb  + (size_t)bh * S_LEN * HDIM;
  const unsigned short* vp = vtb + (size_t)bh * HDIM * S_LEN;

  __shared__ __align__(16) unsigned short ktile[2][32 * 128];  // 2 x 8KB
  __shared__ __align__(16) unsigned short vtile[2][128 * 32];  // 2 x 8KB
  __shared__ __align__(16) unsigned short pls[4][16 * 32];     // per-wave P tile
  unsigned short* myp = pls[wave];

  // staging: 128 threads x 4 chunks of 16B for each of K / V^T
  auto stage_kv = [&](int buf, int kv0) {
#pragma unroll
    for (int j = 0; j < 4; ++j) {
      int c = t + 128 * j;               // 0..511
      int krow = c >> 4;                 // 0..31
      int koff = (c & 15) * 8;           // 0..120
      async_cp16(&ktile[buf][krow * 128 + koff],
                 kp + (size_t)(kv0 + krow) * HDIM + koff);
      int vrow = c >> 2;                 // 0..127
      int voff = (c & 3) * 8;            // 0..24
      async_cp16(&vtile[buf][vrow * 32 + voff],
                 vp + (size_t)vrow * S_LEN + kv0 + voff);
    }
  };

  // Q resident for the whole kv sweep: 4 A-fragments (16 x 128).
  FragAB qf[4];
#pragma unroll
  for (int kk = 0; kk < 4; ++kk) load_afrag(qf[kk], qp, HDIM, q0, kk * 32, lane);

  FragC acc[8] = {};          // O: 16 x 128 fp32
  float m_r[8], l_r[8];
#pragma unroll
  for (int r = 0; r < 8; ++r) { m_r[r] = -1e30f; l_r[r] = 0.0f; }

  const int col   = lane & 15;
  const int rbase = (lane & 16) ? 8 : 0;

  stage_kv(0, 0);

  for (int kv0 = 0; kv0 < S_LEN; kv0 += 32) {
    const int buf = (kv0 >> 5) & 1;
    wait_async_all();
    __syncthreads();                        // K/V tile[buf] visible to all waves
    if (kv0 + 32 < S_LEN) stage_kv(buf ^ 1, kv0 + 32);

    // dots = Q (16x128) @ K^T (128x32): two 16x16 N-tiles, 4-deep K chain.
    FragC d0 = {}, d1 = {};
#pragma unroll
    for (int kk = 0; kk < 4; ++kk) {
      FragAB bf0; load_bfrag_lds(bf0, ktile[buf], 128, 0,  kk * 32, lane);
      d0.v = wmma_bf16(qf[kk].v, bf0.v, d0.v);
      FragAB bf1; load_bfrag_lds(bf1, ktile[buf], 128, 16, kk * 32, lane);
      d1.v = wmma_bf16(qf[kk].v, bf1.v, d1.v);
    }

    // Online softmax. Row stats reduce over the 16 lanes of each half
    // (xor masks 1,2,4,8 never cross the half boundary).
#pragma unroll
    for (int r = 0; r < 8; ++r) {
      float a0 = d0.f[r] * ATT_SCALE;
      float a1 = d1.f[r] * ATT_SCALE;
      float mx = fmaxf(a0, a1);
      mx = fmaxf(mx, __shfl_xor(mx, 1, 32));
      mx = fmaxf(mx, __shfl_xor(mx, 2, 32));
      mx = fmaxf(mx, __shfl_xor(mx, 4, 32));
      mx = fmaxf(mx, __shfl_xor(mx, 8, 32));
      float mn    = fmaxf(m_r[r], mx);
      float alpha = __expf(m_r[r] - mn);
      float p0 = __expf(a0 - mn);
      float p1 = __expf(a1 - mn);
      float sum = p0 + p1;
      sum += __shfl_xor(sum, 1, 32);
      sum += __shfl_xor(sum, 2, 32);
      sum += __shfl_xor(sum, 4, 32);
      sum += __shfl_xor(sum, 8, 32);
      l_r[r] = l_r[r] * alpha + sum;
      m_r[r] = mn;
#pragma unroll
      for (int dt = 0; dt < 8; ++dt) acc[dt].f[r] *= alpha;
      // C-layout -> row-major LDS (bf16) for the A-layout reload.
      int row = r + rbase;
      myp[row * 32 + col]      = f32_to_bf16(p0);
      myp[row * 32 + col + 16] = f32_to_bf16(p1);
    }

    asm volatile("s_wait_dscnt 0x0" ::: "memory");   // P stores visible before reload

    // Reload P (16x32) in A-fragment layout from LDS (ld = 32).
    FragAB pf;
    {
      const int rr   = lane & 15;
      const int koff = (lane & 16) ? 8 : 0;
      const unsigned short* p = myp + rr * 32 + koff;
      pf.q[0] = *(const u32x4*)(p);
      pf.q[1] = *(const u32x4*)(p + 16);
    }

    // O += P (16x32) @ V (32x128); V^T tile makes B-fragments contiguous.
#pragma unroll
    for (int dt = 0; dt < 8; ++dt) {
      FragAB vf; load_bfrag_lds(vf, vtile[buf], 32, dt * 16, 0, lane);
      acc[dt].v = wmma_bf16(pf.v, vf.v, acc[dt].v);
    }
    __syncthreads();                        // all waves done reading tile[buf]
  }

  // finalize: O /= l, write bf16 attn_out [B,S,D] with head interleave.
#pragma unroll
  for (int r = 0; r < 8; ++r) {
    float inv = 1.0f / l_r[r];
    int srow = q0 + r + rbase;
    size_t base = ((size_t)b * S_LEN + srow) * DMODEL + (size_t)h * HDIM;
#pragma unroll
    for (int dt = 0; dt < 8; ++dt) {
      ao[base + dt * 16 + col] = f32_to_bf16(acc[dt].f[r] * inv);
    }
  }
}

// ---- kernel 4: output projection + bias, fp32 out -----------------------
__global__ __launch_bounds__(256) void gemm_out_kernel(
    const unsigned short* __restrict__ ao, const unsigned short* __restrict__ woutT,
    const float* __restrict__ b_out, float* __restrict__ out) {
  const int t    = threadIdx.x;
  const int lane = t & 31;
  const int wave = t >> 5;
  const int m0 = blockIdx.x * 128 + wave * 16;
  const int n0 = blockIdx.y * 64;

  __shared__ __align__(16) unsigned short btile[2][64 * 32];

  stage_btile(btile[0], woutT, DMODEL, n0, 0, t);

  FragC c[4] = {};
  for (int k0 = 0; k0 < DMODEL; k0 += 32) {
    const int buf = (k0 >> 5) & 1;
    wait_async_all();
    __syncthreads();
    if (k0 + 32 < DMODEL)
      stage_btile(btile[buf ^ 1], woutT, DMODEL, n0, k0 + 32, t);

    FragAB af; load_afrag(af, ao, DMODEL, m0, k0, lane);
#pragma unroll
    for (int nt = 0; nt < 4; ++nt) {
      FragAB bf; load_bfrag_lds(bf, btile[buf], 32, nt * 16, 0, lane);
      c[nt].v = wmma_bf16(af.v, bf.v, c[nt].v);
    }
    __syncthreads();
  }

  const int col   = lane & 15;
  const int rbase = (lane & 16) ? 8 : 0;
#pragma unroll
  for (int nt = 0; nt < 4; ++nt) {
#pragma unroll
    for (int r = 0; r < 8; ++r) {
      int m = m0 + rbase + r;
      int n = n0 + nt * 16 + col;
      out[(size_t)m * DMODEL + n] = c[nt].f[r] + b_out[n];
    }
  }
}

// ---- host-side launch ---------------------------------------------------
extern "C" void kernel_launch(void* const* d_in, const int* in_sizes, int n_in,
                              void* d_out, int out_size, void* d_ws, size_t ws_size,
                              hipStream_t stream) {
  (void)in_sizes; (void)n_in; (void)out_size; (void)ws_size;
  const float* x     = (const float*)d_in[0];
  const float* w_qkv = (const float*)d_in[1];
  const float* w_out = (const float*)d_in[2];
  const float* b_out = (const float*)d_in[3];
  float* out = (float*)d_out;

  char* ws = (char*)d_ws;
  size_t o = 0;
  unsigned short* xb    = (unsigned short*)(ws + o); o += (size_t)MROWS * DMODEL * 2;
  unsigned short* wqkvT = (unsigned short*)(ws + o); o += (size_t)N_QKV * DMODEL * 2;
  unsigned short* woutT = (unsigned short*)(ws + o); o += (size_t)DMODEL * DMODEL * 2;
  unsigned short* qb    = (unsigned short*)(ws + o); o += (size_t)BH * S_LEN * HDIM * 2;
  unsigned short* kb    = (unsigned short*)(ws + o); o += (size_t)BH * S_LEN * HDIM * 2;
  unsigned short* vtb   = (unsigned short*)(ws + o); o += (size_t)BH * HDIM * S_LEN * 2;
  unsigned short* ao    = (unsigned short*)(ws + o); o += (size_t)MROWS * DMODEL * 2;

  {
    int n = MROWS * DMODEL;
    cast_bf16_kernel<<<(n + 255) / 256, 256, 0, stream>>>(x, xb, n);
  }
  {
    long t = (long)N_QKV * DMODEL;
    transpose_cast_kernel<<<(int)((t + 255) / 256), 256, 0, stream>>>(
        w_qkv, wqkvT, DMODEL, N_QKV);
  }
  {
    long t = (long)DMODEL * DMODEL;
    transpose_cast_kernel<<<(int)((t + 255) / 256), 256, 0, stream>>>(
        w_out, woutT, DMODEL, DMODEL);
  }
  gemm_qkv_kernel<<<dim3(MROWS / 128, N_QKV / 64), 256, 0, stream>>>(
      xb, wqkvT, qb, kb, vtb);
  flash_attn_kernel<<<dim3(S_LEN / 64, BH), 128, 0, stream>>>(qb, kb, vtb, ao);
  gemm_out_kernel<<<dim3(MROWS / 128, DMODEL / 64), 256, 0, stream>>>(
      ao, woutT, b_out, out);
}
